// PraxisNano_4294967296388
// MI455X (gfx1250) — matrix operand order repaired and checked by reading, not compiled, not run on hardware
//
#include <hip/hip_runtime.h>
#include <hip/hip_bf16.h>

typedef __attribute__((ext_vector_type(16))) __bf16 v16bf;
typedef __attribute__((ext_vector_type(8)))  float  v8f;

#define TT 8192
#define HH 512
#define EE 2048
#define BBATCH 4
#define CC 64
#define CHUNK_STRIDE 32
#define NCHUNK (TT / CHUNK_STRIDE)   // 256
#define THREADS 512                  // 16 waves (wave32)
#define LN_EPS 1e-5f

// ---------- helpers ----------

static __device__ __forceinline__ unsigned short f2bf(float f) {
  unsigned u = __builtin_bit_cast(unsigned, f);
  // round-to-nearest-even bf16
  return (unsigned short)((u + 0x7fffu + ((u >> 16) & 1u)) >> 16);
}

union FragU { v16bf v; uint4 q[2]; };
union AccU  { v8f  v; float f[8]; };

// A fragment (16x32 bf16, M x K) from M-major memory (K contiguous, leading dim ld).
// Per ISA 7.12.2: lane L (r=L&15,g=L>>4) holds row M=r, K = g*8+{0..7} and g*8+{16..23}.
static __device__ __forceinline__ v16bf ldfragA(const unsigned short* p, int ld,
                                                int row0, int col0, int lane) {
  int r = lane & 15, g = lane >> 4;
  const unsigned short* q = p + (size_t)(row0 + r) * ld + col0 + g * 8;
  FragU f;
  f.q[0] = *(const uint4*)(q);
  f.q[1] = *(const uint4*)(q + 16);
  return f.v;
}

// B fragment (32x16 bf16, K x N) from N-major memory (K contiguous, leading dim ld).
// Lane L holds column N=L&15, K = (L>>4)*8+{0..7} and (L>>4)*8+{16..23}.
static __device__ __forceinline__ v16bf ldfragB(const unsigned short* p, int ld,
                                                int row0, int col0, int lane) {
  int n = lane & 15, g = lane >> 4;
  const unsigned short* q = p + (size_t)(row0 + n) * ld + col0 + g * 8;
  FragU f;
  f.q[0] = *(const uint4*)(q);
  f.q[1] = *(const uint4*)(q + 16);
  return f.v;
}

static __device__ __forceinline__ v8f wmma_bf16(v16bf a, v16bf b, v8f c) {
  return __builtin_amdgcn_wmma_f32_16x16x32_bf16(false, a, false, b, (short)0, c,
                                                 false, false);
}

// ---------- kernel 0: weights -> bf16 transposed (workspace) ----------
// ws layout: [0, E*H)  = fw1^T (E x H) bf16 ;  [E*H, 2*E*H) = fw2^T (H x E) bf16
__global__ __launch_bounds__(256)
void praxis_prep_weights(const float* __restrict__ fw1, const float* __restrict__ fw2,
                         unsigned short* __restrict__ ws) {
  int idx = blockIdx.x * 256 + threadIdx.x;
  if (idx < EE * HH) {
    int e = idx / HH, h = idx - e * HH;
    ws[idx] = f2bf(fw1[(size_t)h * EE + e]);                 // fw1T[e][h]
    int h2 = idx / EE, e2 = idx - h2 * EE;
    ws[EE * HH + idx] = f2bf(fw2[(size_t)e2 * HH + h2]);     // fw2T[h][e]
  }
}

// ---------- kernel 1: zero the accumulation output ----------
__global__ __launch_bounds__(256)
void praxis_zero_out(float4* __restrict__ out, int n4) {
  int i = blockIdx.x * 256 + threadIdx.x;
  if (i < n4) out[i] = make_float4(0.f, 0.f, 0.f, 0.f);
}

// ---------- kernel 2: fused per-chunk pipeline ----------
__global__ __launch_bounds__(THREADS)
void praxis_chunk(const float* __restrict__ x,
                  const float* __restrict__ ln1_g, const float* __restrict__ ln1_b,
                  const float* __restrict__ W1,    const float* __restrict__ W2,
                  const float* __restrict__ ln2_g, const float* __restrict__ ln2_b,
                  const unsigned short* __restrict__ fw1T, const float* __restrict__ fb1,
                  const unsigned short* __restrict__ fw2T, const float* __restrict__ fb2,
                  float* __restrict__ out) {
  extern __shared__ char smem[];
  float*          sRes = (float*)smem;                                   // 64x512 f32 (resid -> c2)
  unsigned short* sT   = (unsigned short*)(smem + 131072);               // 512x64 cnT/yT, later 64x512 hn
  unsigned short* sU   = (unsigned short*)(smem + 131072 + 65536);       // 64x128 bf16 U slab
  unsigned short* sW1  = (unsigned short*)(smem + 131072 + 65536 + 16384); // 64x64 bf16
  unsigned short* sW2  = sW1 + CC * CC;

  const int bid = blockIdx.x;
  const int b = bid / NCHUNK, nch = bid - b * NCHUNK;
  const int start = nch * CHUNK_STRIDE;
  const int tid = threadIdx.x, lane = tid & 31, wid = tid >> 5;
  const int mt = wid & 3, ng = wid >> 2;     // stage-B / TriLinear tiling: wave -> (mt, 8 n-tiles)
  const int lc = lane & 15, lg = lane >> 4;  // C/D layout: n = lc, m = 8*lg + v

  // --- phase 0: masked chunk load + tri-masked bf16 weights ---
  for (int i = tid; i < CC * HH / 4; i += THREADS) {
    int r = i >> 7, cv = i & 127;
    int pos = start + r;
    float4 v = make_float4(0.f, 0.f, 0.f, 0.f);
    if (pos < TT) v = *(const float4*)(x + ((size_t)b * TT + pos) * HH + cv * 4);
    *(float4*)(sRes + r * HH + cv * 4) = v;
  }
  for (int i = tid; i < CC * CC; i += THREADS) {
    int t = i >> 6, s = i & 63;
    sW1[i] = (s <= t) ? f2bf(W1[i]) : (unsigned short)0;
    sW2[i] = (s <= t) ? f2bf(W2[i]) : (unsigned short)0;
  }
  // warm L2 for the first FFW slabs while we do LN/TriLinear
  __builtin_prefetch(fw1T + (size_t)tid * 256, 0, 1);
  __builtin_prefetch(fw2T + (size_t)tid * EE, 0, 1);
  __syncthreads();

  // --- phase 1: LN1 -> cnT (transposed bf16 [h][t], ld=64) ---
  for (int i = 0; i < 4; i++) {
    int r = wid * 4 + i;
    float vals[16], s = 0.f, s2 = 0.f;
    for (int k = 0; k < 16; k++) {
      float v = sRes[r * HH + lane + 32 * k];
      vals[k] = v; s += v; s2 += v * v;
    }
    for (int o = 16; o > 0; o >>= 1) { s += __shfl_xor(s, o, 32); s2 += __shfl_xor(s2, o, 32); }
    float m = s * (1.f / HH);
    float rstd = rsqrtf(s2 * (1.f / HH) - m * m + LN_EPS);
    for (int k = 0; k < 16; k++) {
      int h = lane + 32 * k;
      sT[h * CC + r] = f2bf((vals[k] - m) * rstd * ln1_g[h] + ln1_b[h]);
    }
  }
  __syncthreads();

  // --- phase 2: y = (W1*tri) @ cn  (A = W1 LDS, B = cnT LDS) ---
  // A tile (mt, kt) is invariant across the 8 j tiles: hoist it (kt outer, j inner).
  v8f yacc[8];
  v8f vzero = {};
  for (int j = 0; j < 8; j++) yacc[j] = vzero;
  for (int kt = 0; kt < 2; kt++) {
    v16bf a = ldfragA(sW1, CC, mt * 16, kt * 32, lane);
    for (int j = 0; j < 8; j++) {
      v16bf bb = ldfragB(sT, CC, (ng * 8 + j) * 16, kt * 32, lane);
      yacc[j] = wmma_bf16(a, bb, yacc[j]);
    }
  }
  __syncthreads();                      // all cnT reads done
  for (int j = 0; j < 8; j++) {         // store yT over cnT: per lane 8 contiguous m
    int nt = ng * 8 + j;
    unsigned short tmp[8];
    AccU au; au.v = yacc[j];
    for (int v = 0; v < 8; v++) tmp[v] = f2bf(au.f[v]);
    *(uint4*)(sT + (size_t)(nt * 16 + lc) * CC + mt * 16 + lg * 8) = *(const uint4*)tmp;
  }
  __syncthreads();

  // --- phase 3: y2 = (W2*tri) @ y ---
  for (int j = 0; j < 8; j++) yacc[j] = vzero;
  for (int kt = 0; kt < 2; kt++) {
    v16bf a = ldfragA(sW2, CC, mt * 16, kt * 32, lane);
    for (int j = 0; j < 8; j++) {
      v16bf bb = ldfragB(sT, CC, (ng * 8 + j) * 16, kt * 32, lane);
      yacc[j] = wmma_bf16(a, bb, yacc[j]);
    }
  }

  // --- phase 4: c2 = y2 + resid (in place, element-exclusive) ---
  for (int j = 0; j < 8; j++) {
    int nt = ng * 8 + j;
    AccU au; au.v = yacc[j];
    for (int v = 0; v < 8; v++) {
      int m = mt * 16 + lg * 8 + v;
      sRes[m * HH + nt * 16 + lc] += au.f[v];
    }
  }
  __syncthreads();

  // --- phase 5: LN2 -> hn (row-major bf16 [t][h], reuses sT) ---
  unsigned short* hn = sT;
  for (int i = 0; i < 4; i++) {
    int r = wid * 4 + i;
    float vals[16], s = 0.f, s2 = 0.f;
    for (int k = 0; k < 16; k++) {
      float v = sRes[r * HH + lane + 32 * k];
      vals[k] = v; s += v; s2 += v * v;
    }
    for (int o = 16; o > 0; o >>= 1) { s += __shfl_xor(s, o, 32); s2 += __shfl_xor(s2, o, 32); }
    float m = s * (1.f / HH);
    float rstd = rsqrtf(s2 * (1.f / HH) - m * m + LN_EPS);
    for (int k = 0; k < 16; k++) {
      int h = lane + 32 * k;
      hn[r * HH + h] = f2bf((vals[k] - m) * rstd * ln2_g[h] + ln2_b[h]);
    }
  }
  __syncthreads();

  // --- phase 6: FFW, blocked over E in 128-wide slabs ---
  v8f facc[8];
  for (int j = 0; j < 8; j++) facc[j] = vzero;
  const int mtU = wid & 3, ntU0 = (wid >> 2) * 2;   // stage-A: 4x8 tiles, 2 per wave

  for (int eb = 0; eb < EE; eb += 128) {
    // prefetch the next slab of both weight streams into L2/L0 while computing
    if (eb + 128 < EE) {
      __builtin_prefetch(fw1T + (size_t)(eb + 128) * HH + (size_t)tid * 128, 0, 1);
      __builtin_prefetch(fw2T + (size_t)tid * EE + (eb + 128), 0, 1);
    }
    // stage A: U = sin(hn @ fw1[:,eb:eb+128] + fb1)
    // A fragment shared by both u tiles: load once per kt.
    v8f uacc0 = vzero, uacc1 = vzero;
    const unsigned short* b1 = fw1T + (size_t)eb * HH;
    for (int kt = 0; kt < 16; kt++) {
      v16bf a   = ldfragA(hn, HH, mtU * 16, kt * 32, lane);
      v16bf bb0 = ldfragB(b1, HH, (ntU0 + 0) * 16, kt * 32, lane);
      v16bf bb1 = ldfragB(b1, HH, (ntU0 + 1) * 16, kt * 32, lane);
      uacc0 = wmma_bf16(a, bb0, uacc0);
      uacc1 = wmma_bf16(a, bb1, uacc1);
    }
    __syncthreads();                  // prior stage-B done reading sU
    for (int u = 0; u < 2; u++) {
      int ntu = ntU0 + u;
      float bias = fb1[eb + ntu * 16 + lc];
      AccU au; au.v = (u == 0) ? uacc0 : uacc1;
      for (int v = 0; v < 8; v++) {
        int m = mtU * 16 + lg * 8 + v;
        sU[m * 128 + ntu * 16 + lc] = f2bf(__sinf(au.f[v] + bias));
      }
    }
    __syncthreads();                  // U slab visible
    // stage B: ACC += U @ fw2[eb:eb+128, :]
    // A fragment (from sU) shared by all 8 j tiles: load once per kt.
    for (int kt = 0; kt < 4; kt++) {
      v16bf a = ldfragA(sU, 128, mt * 16, kt * 32, lane);
      for (int j = 0; j < 8; j++) {
        v16bf bb = ldfragB(fw2T + eb, EE, (ng * 8 + j) * 16, kt * 32, lane);
        facc[j] = wmma_bf16(a, bb, facc[j]);
      }
    }
  }

  // --- phase 7: out_chunks = (ffw + fb2 + c2) * mask; scatter-add ---
  for (int j = 0; j < 8; j++) {
    int nt = ng * 8 + j;
    int h = nt * 16 + lc;
    float bias = fb2[h];
    AccU au; au.v = facc[j];
    for (int v = 0; v < 8; v++) {
      int m = mt * 16 + lg * 8 + v;
      int pos = start + m;
      if (pos < TT) {
        float val = au.f[v] + bias + sRes[m * HH + h];
        atomicAdd(out + ((size_t)b * TT + pos) * HH + h, val);
      }
    }
  }
}

// ---------- kernel 3: divide by coverage counts (analytic: 1 if t<32 else 2) ----------
__global__ __launch_bounds__(256)
void praxis_scale(float4* __restrict__ out, int n4) {
  int i = blockIdx.x * 256 + threadIdx.x;
  if (i < n4) {
    size_t e = (size_t)i * 4;
    int t = (int)((e / HH) % TT);
    float f = (t < CHUNK_STRIDE) ? 1.f : 0.5f;
    float4 v = out[i];
    v.x *= f; v.y *= f; v.z *= f; v.w *= f;
    out[i] = v;
  }
}

// ---------- launcher ----------
extern "C" void kernel_launch(void* const* d_in, const int* in_sizes, int n_in,
                              void* d_out, int out_size, void* d_ws, size_t ws_size,
                              hipStream_t stream) {
  (void)in_sizes; (void)n_in; (void)out_size; (void)ws_size;
  const float* x     = (const float*)d_in[0];
  const float* ln1_g = (const float*)d_in[1];
  const float* ln1_b = (const float*)d_in[2];
  const float* W1    = (const float*)d_in[3];
  const float* W2    = (const float*)d_in[4];
  const float* ln2_g = (const float*)d_in[5];
  const float* ln2_b = (const float*)d_in[6];
  const float* fw1   = (const float*)d_in[7];
  const float* fb1   = (const float*)d_in[8];
  const float* fw2   = (const float*)d_in[9];
  const float* fb2   = (const float*)d_in[10];
  float* out = (float*)d_out;

  unsigned short* wsbf = (unsigned short*)d_ws;      // fw1T (E*H) then fw2T (H*E), bf16
  const unsigned short* fw1T = wsbf;
  const unsigned short* fw2T = wsbf + (size_t)EE * HH;

  const int n4 = BBATCH * TT * HH / 4;

  praxis_prep_weights<<<(EE * HH + 255) / 256, 256, 0, stream>>>(fw1, fw2, wsbf);
  praxis_zero_out<<<(n4 + 255) / 256, 256, 0, stream>>>((float4*)out, n4);

  const size_t lds = 131072 + 65536 + 16384 + 16384; // 224 KB
  praxis_chunk<<<BBATCH * NCHUNK, THREADS, lds, stream>>>(
      x, ln1_g, ln1_b, W1, W2, ln2_g, ln2_b, fw1T, fb1, fw2T, fb2, out);

  praxis_scale<<<(n4 + 255) / 256, 256, 0, stream>>>((float4*)out, n4);
}
